// StandardTransformer27M_82729660056337
// MI455X (gfx1250) — matrix operand order repaired and checked by reading, not compiled, not run on hardware
//
#include <hip/hip_runtime.h>
#include <math.h>

// Problem constants (fixed by the reference)
#define BB   2
#define NN   2048
#define LL   6
#define HH   8
#define DD   400
#define FFF  1600
#define VV   32000
#define HDQ  50          // true head dim
#define HDP  64          // padded head dim for WMMA
#define BN   (BB*NN)     // 4096 tokens
#define BMT  128         // GEMM M tile
#define BNT  64          // GEMM N tile
#define KP1  416         // ceil(400/32)*32 : padded K for D=400 activations

typedef __attribute__((ext_vector_type(16))) _Float16 v16h;
typedef __attribute__((ext_vector_type(8)))  _Float16 h8v;   // 16-byte unit
typedef __attribute__((ext_vector_type(8)))  float    v8f;

static __device__ __forceinline__ v8f wmma32(v16h a, v16h b, v8f c) {
  // D = A(16x32 f16) * B(32x16 f16) + C(16x16 f32)
  return __builtin_amdgcn_wmma_f32_16x16x32_f16(false, a, false, b, (short)0, c,
                                                false, false);
}

// ---------------------------------------------------------------------------
// Embedding: x[b,n,:] = tok_emb[idx[b,n],:] + pos_emb[n,:]   (fp32)
// ---------------------------------------------------------------------------
__global__ __launch_bounds__(256)
void k_embed(const int* __restrict__ idx, const float* __restrict__ tok,
             const float* __restrict__ pos, float* __restrict__ x) {
  int total = BN * DD;
  for (int i = blockIdx.x * blockDim.x + threadIdx.x; i < total;
       i += gridDim.x * blockDim.x) {
    int row = i / DD, d = i - row * DD;
    int n = row % NN;
    x[i] = tok[(size_t)idx[row] * DD + d] + pos[(size_t)n * DD + d];
  }
}

// ---------------------------------------------------------------------------
// LayerNorm over D=400; writes f16 rows with stride KP1 (pad left as-is)
// ---------------------------------------------------------------------------
__global__ __launch_bounds__(256)
void k_ln(const float* __restrict__ x, const float* __restrict__ w,
          const float* __restrict__ b, _Float16* __restrict__ y) {
  __shared__ float sbuf[256], ssbuf[256];
  int t = threadIdx.x;
  size_t row = blockIdx.x;
  const float* xr = x + row * DD;
  float s = 0.f, ss = 0.f;
  for (int d = t; d < DD; d += 256) { float v = xr[d]; s += v; ss += v * v; }
  sbuf[t] = s; ssbuf[t] = ss;
  __syncthreads();
  for (int off = 128; off > 0; off >>= 1) {
    if (t < off) { sbuf[t] += sbuf[t + off]; ssbuf[t] += ssbuf[t + off]; }
    __syncthreads();
  }
  float mean = sbuf[0] * (1.0f / DD);
  float var  = ssbuf[0] * (1.0f / DD) - mean * mean;
  float inv  = rsqrtf(var + 1e-5f);
  for (int d = t; d < DD; d += 256)
    y[row * KP1 + d] = (_Float16)((xr[d] - mean) * inv * w[d] + b[d]);
}

// ---------------------------------------------------------------------------
// Pack fp32 weights B[K][N] -> transposed, zero-padded f16 Bt[Npad][Kpad].
// One-shot per GEMM; clamped unconditional reads.
// ---------------------------------------------------------------------------
__global__ __launch_bounds__(256)
void k_pack_b(const float* __restrict__ B, _Float16* __restrict__ Bt,
              int K, int N, int Kpad, int Npad) {
  int total = Npad * Kpad;
  for (int i = blockIdx.x * blockDim.x + threadIdx.x; i < total;
       i += gridDim.x * blockDim.x) {
    int n = i / Kpad, k = i - n * Kpad;
    int kc = k < K ? k : K - 1;
    int nc = n < N ? n : N - 1;
    float v = B[(size_t)kc * N + nc];
    Bt[i] = (_Float16)((k < K && n < N) ? v : 0.f);
  }
}

// ---------------------------------------------------------------------------
// WMMA GEMM: C[M,N] = epilogue(A[M,Kpad](f16) * Bt[Npad,Kpad](f16) + bias)
// OP: 0 = bias->Cf, 1 = bias+residual->Cf, 2 = gelu(bias)->Ch (f16, stride N)
// Block = 256 threads (8 waves), tile 128x64, K-step 32.
// Software-pipelined: double-buffered LDS, one barrier per K-step, next tile
// held in registers during WMMA. Staging is guard-free (Bt pads are zero).
// ---------------------------------------------------------------------------
template <int OP>
__global__ __launch_bounds__(256)
void k_gemm(const _Float16* __restrict__ A, int lda,
            const _Float16* __restrict__ Bt, int ldb,
            const float* __restrict__ bias, const float* __restrict__ res,
            float* __restrict__ Cf, _Float16* __restrict__ Ch,
            int N, int Kt) {
  __shared__ _Float16 As[2][BMT * 32];   // As[buf][m][k]
  __shared__ _Float16 Bs[2][BNT * 32];   // Bs[buf][n][k]
  int tid = threadIdx.x;
  int wave = tid >> 5, lane = tid & 31;
  int bm = blockIdx.y, bn = blockIdx.x;
  int wr = wave & 3;          // wave M row (covers 2 x 16 rows)
  int wc = wave >> 2;         // wave N half (covers 2 x 16 cols)
  v8f acc[2][2];
  acc[0][0] = (v8f){}; acc[0][1] = (v8f){};
  acc[1][0] = (v8f){}; acc[1][1] = (v8f){};
  const _Float16* Abase = A + (size_t)bm * BMT * lda;
  const _Float16* Bbase = Bt + (size_t)bn * BNT * ldb;
  int ar = tid >> 2;                  // rows 0..63 (and +64 for A)
  int kc = (tid & 3) * 8;             // 8-half chunk within 32-wide k slice
  // preload K-tile 0 into registers
  h8v ga0 = *(const h8v*)(Abase + (size_t)ar * lda + kc);
  h8v ga1 = *(const h8v*)(Abase + (size_t)(ar + 64) * lda + kc);
  h8v gb  = *(const h8v*)(Bbase + (size_t)ar * ldb + kc);
  for (int kt = 0; kt < Kt; ++kt) {
    int buf = kt & 1;
    *(h8v*)&As[buf][ar * 32 + kc]        = ga0;
    *(h8v*)&As[buf][(ar + 64) * 32 + kc] = ga1;
    *(h8v*)&Bs[buf][ar * 32 + kc]        = gb;
    __syncthreads();
    if (kt + 1 < Kt) {
      int k0 = (kt + 1) * 32;
      ga0 = *(const h8v*)(Abase + (size_t)ar * lda + k0 + kc);
      ga1 = *(const h8v*)(Abase + (size_t)(ar + 64) * lda + k0 + kc);
      gb  = *(const h8v*)(Bbase + (size_t)ar * ldb + k0 + kc);
      // speculative prefetch of the tile after next (dropped if OOB)
      __builtin_prefetch(Abase + (size_t)ar * lda + k0 + 32 + kc, 0, 1);
    }
    // A fragments: lane=row, kbase=(lane>>4)*8; elems 0..7 -> kb..kb+7,
    // elems 8..15 -> kb+16..kb+23 (16B-aligned b128 LDS loads)
    v16h a[2], b[2];
#pragma unroll
    for (int mt = 0; mt < 2; ++mt) {
      int row = (wr * 2 + mt) * 16 + (lane & 15);
      const h8v* p = (const h8v*)&As[buf][row * 32 + (lane >> 4) * 8];
      h8v lo = p[0], hi = p[2];
#pragma unroll
      for (int e = 0; e < 8; ++e) { a[mt][e] = lo[e]; a[mt][e + 8] = hi[e]; }
    }
    // B fragments: lane=col, kbase=(lane>>4)*8; elems 0..7 -> kb..kb+7,
    // elems 8..15 -> kb+16..kb+23
#pragma unroll
    for (int nt = 0; nt < 2; ++nt) {
      int col = (wc * 2 + nt) * 16 + (lane & 15);
      const h8v* p = (const h8v*)&Bs[buf][col * 32 + (lane >> 4) * 8];
      h8v lo = p[0], hi = p[2];
#pragma unroll
      for (int e = 0; e < 8; ++e) { b[nt][e] = lo[e]; b[nt][e + 8] = hi[e]; }
    }
#pragma unroll
    for (int mt = 0; mt < 2; ++mt)
#pragma unroll
      for (int nt = 0; nt < 2; ++nt)
        acc[mt][nt] = wmma32(a[mt], b[nt], acc[mt][nt]);
  }
  // Epilogue. D layout: col = lane%16, row = (lane>>4)*8 + elem
  // M (=BN) is an exact multiple of BMT, so no row guard.
#pragma unroll
  for (int mt = 0; mt < 2; ++mt) {
    int rbase = bm * BMT + (wr * 2 + mt) * 16 + (lane >> 4) * 8;
#pragma unroll
    for (int nt = 0; nt < 2; ++nt) {
      int col = bn * BNT + (wc * 2 + nt) * 16 + (lane & 15);
      if (col >= N) continue;
      float bv = bias ? bias[col] : 0.f;
#pragma unroll
      for (int e = 0; e < 8; ++e) {
        size_t r = (size_t)(rbase + e);
        float v = acc[mt][nt][e] + bv;
        if constexpr (OP == 2) {
          v = 0.5f * v * (1.0f + erff(v * 0.70710678f));
          Ch[r * N + col] = (_Float16)v;
        } else {
          if constexpr (OP == 1) v += res[r * N + col];
          Cf[r * N + col] = v;
        }
      }
    }
  }
}

// ---------------------------------------------------------------------------
// Split QKV [BN,1200] into padded f16 Q/K [B,H,N,64] and transposed V [B,H,64,N]
// ---------------------------------------------------------------------------
__global__ __launch_bounds__(256)
void k_qkv_split(const float* __restrict__ qkv, _Float16* __restrict__ Qp,
                 _Float16* __restrict__ Kp, _Float16* __restrict__ Vt) {
  int total = BB * HH * NN * HDP;
  for (int i = blockIdx.x * blockDim.x + threadIdx.x; i < total;
       i += gridDim.x * blockDim.x) {
    int c = i & (HDP - 1);
    int n = (i >> 6) % NN;
    int bh = i / (HDP * NN);
    int b = bh / HH, h = bh % HH;
    const float* base = qkv + (size_t)(b * NN + n) * (3 * DD);
    int cc = c < HDQ ? c : HDQ - 1;
    float qv = base[h * HDQ + cc];
    float kv = base[DD + h * HDQ + cc];
    float vv = base[2 * DD + h * HDQ + cc];
    bool ok = c < HDQ;
    Qp[i] = (_Float16)(ok ? qv : 0.f);
    Kp[i] = (_Float16)(ok ? kv : 0.f);
    Vt[((size_t)bh * HDP + c) * NN + n] = (_Float16)(ok ? vv : 0.f);
  }
}

// ---------------------------------------------------------------------------
// Flash attention, WMMA, causal. 1 wave = 16 queries of one (b,h).
// Q/K: [B,H,N,64] f16 (zero-padded); Vt: [B,H,64,N] f16.
// Out: f16 [BN][KP1] rows (col = h*50+ch); pad columns left untouched
// (they multiply zeroed Bt pad rows in the next GEMM).
// ---------------------------------------------------------------------------
__global__ __launch_bounds__(128)
void k_attn(const _Float16* __restrict__ Qp, const _Float16* __restrict__ Kp,
            const _Float16* __restrict__ Vt, _Float16* __restrict__ Out) {
  __shared__ _Float16 Plds[4][16 * 32];   // per-wave P scratch (16 q x 32 keys)
  const float NEG = -3.0e38f;
  const float scale = 0.141421356f;       // 1/sqrt(50)
  int wave = threadIdx.x >> 5, lane = threadIdx.x & 31;
  int gt = blockIdx.x * 4 + wave;
  int tilesPerBH = NN >> 4;
  int bh = gt / tilesPerBH;
  int qbase = (gt % tilesPerBH) << 4;
  int b = bh / HH, h = bh % HH;
  const _Float16* Q  = Qp + (size_t)bh * NN * HDP;
  const _Float16* Kb = Kp + (size_t)bh * NN * HDP;
  const _Float16* V  = Vt + (size_t)bh * HDP * NN;
  int kb8 = (lane >> 4) * 8;
  int kk16 = (lane >> 4) * 16;
  v16h a_lo, a_hi;
  {
    const h8v* qp = (const h8v*)(Q + (size_t)(qbase + (lane & 15)) * HDP + kk16);
    h8v q0 = qp[0], q1 = qp[1], q2 = qp[4], q3 = qp[5];
#pragma unroll
    for (int e = 0; e < 8; ++e) {
      a_lo[e] = q0[e]; a_lo[e + 8] = q1[e];
      a_hi[e] = q2[e]; a_hi[e + 8] = q3[e];
    }
  }
  v8f o[4]; o[0] = (v8f){}; o[1] = (v8f){}; o[2] = (v8f){}; o[3] = (v8f){};
  float m_i[8], l_i[8];
#pragma unroll
  for (int e = 0; e < 8; ++e) { m_i[e] = NEG; l_i[e] = 0.f; }
  _Float16* P = Plds[wave];
  for (int kb = 0; kb < qbase + 16; kb += 32) {
    v16h bk0l, bk0h, bk1l, bk1h;
    {
      const h8v* kp0 = (const h8v*)(Kb + (size_t)(kb + (lane & 15)) * HDP + kk16);
      const h8v* kp1 = (const h8v*)(Kb + (size_t)(kb + 16 + (lane & 15)) * HDP + kk16);
      h8v k0a = kp0[0], k0b = kp0[1], k0c = kp0[4], k0d = kp0[5];
      h8v k1a = kp1[0], k1b = kp1[1], k1c = kp1[4], k1d = kp1[5];
#pragma unroll
      for (int e = 0; e < 8; ++e) {
        bk0l[e] = k0a[e]; bk0l[e + 8] = k0b[e];
        bk0h[e] = k0c[e]; bk0h[e + 8] = k0d[e];
        bk1l[e] = k1a[e]; bk1l[e + 8] = k1b[e];
        bk1h[e] = k1c[e]; bk1h[e + 8] = k1d[e];
      }
    }
    v8f s0 = {}, s1 = {};
    s0 = wmma32(a_lo, bk0l, s0); s0 = wmma32(a_hi, bk0h, s0);
    s1 = wmma32(a_lo, bk1l, s1); s1 = wmma32(a_hi, bk1h, s1);
    int key0 = kb + (lane & 15);
    int key1 = key0 + 16;
    float p0[8], p1[8], corr[8];
#pragma unroll
    for (int e = 0; e < 8; ++e) {
      int qr = qbase + (lane >> 4) * 8 + e;
      float v0 = (key0 <= qr) ? s0[e] * scale : NEG;
      float v1 = (key1 <= qr) ? s1[e] * scale : NEG;
      float mx = fmaxf(v0, v1);
      for (int d = 1; d < 16; d <<= 1) mx = fmaxf(mx, __shfl_xor(mx, d, 32));
      float m_new = fmaxf(m_i[e], mx);
      corr[e] = expf(m_i[e] - m_new);
      float e0 = expf(v0 - m_new), e1 = expf(v1 - m_new);
      float rs = e0 + e1;
      for (int d = 1; d < 16; d <<= 1) rs += __shfl_xor(rs, d, 32);
      l_i[e] = l_i[e] * corr[e] + rs;
      m_i[e] = m_new;
      p0[e] = e0; p1[e] = e1;
    }
#pragma unroll
    for (int c = 0; c < 4; ++c)
#pragma unroll
      for (int e = 0; e < 8; ++e) o[c][e] *= corr[e];
#pragma unroll
    for (int e = 0; e < 8; ++e) {
      int r = (lane >> 4) * 8 + e;
      P[r * 32 + (lane & 15)]      = (_Float16)p0[e];
      P[r * 32 + 16 + (lane & 15)] = (_Float16)p1[e];
    }
    v16h pa;
    {
      const h8v* pp = (const h8v*)&P[(lane & 15) * 32 + kb8];
      h8v pl = pp[0], ph = pp[2];
#pragma unroll
      for (int e = 0; e < 8; ++e) { pa[e] = pl[e]; pa[e + 8] = ph[e]; }
    }
#pragma unroll
    for (int c = 0; c < 4; ++c) {
      v16h bv;
      const h8v* vp =
          (const h8v*)(V + (size_t)(c * 16 + (lane & 15)) * NN + kb + kb8);
      h8v v0 = vp[0], v1 = vp[2];
#pragma unroll
      for (int e = 0; e < 8; ++e) { bv[e] = v0[e]; bv[e + 8] = v1[e]; }
      o[c] = wmma32(pa, bv, o[c]);
    }
  }
#pragma unroll
  for (int c = 0; c < 4; ++c) {
    int ch = c * 16 + (lane & 15);
    if (ch >= HDQ) continue;
#pragma unroll
    for (int e = 0; e < 8; ++e) {
      int r = qbase + (lane >> 4) * 8 + e;
      Out[(size_t)(b * NN + r) * KP1 + h * HDQ + ch] =
          (_Float16)(o[c][e] / l_i[e]);
    }
  }
}

// ---------------------------------------------------------------------------
// Host orchestration
// ---------------------------------------------------------------------------
extern "C" void kernel_launch(void* const* d_in, const int* in_sizes, int n_in,
                              void* d_out, int out_size, void* d_ws, size_t ws_size,
                              hipStream_t stream) {
  (void)in_sizes; (void)n_in; (void)out_size; (void)ws_size;
  const int*   idx   = (const int*)  d_in[0];
  const float* tok   = (const float*)d_in[1];
  const float* pos   = (const float*)d_in[2];
  const float* ln1w  = (const float*)d_in[3];
  const float* ln1b  = (const float*)d_in[4];
  const float* qkvw  = (const float*)d_in[5];
  const float* qkvb  = (const float*)d_in[6];
  const float* aow   = (const float*)d_in[7];
  const float* aob   = (const float*)d_in[8];
  const float* ln2w  = (const float*)d_in[9];
  const float* ln2b  = (const float*)d_in[10];
  const float* fc1w  = (const float*)d_in[11];
  const float* fc1b  = (const float*)d_in[12];
  const float* fc2w  = (const float*)d_in[13];
  const float* fc2b  = (const float*)d_in[14];
  const float* lnfw  = (const float*)d_in[15];
  const float* lnfb  = (const float*)d_in[16];
  const float* headw = (const float*)d_in[17];

  char* w = (char*)d_ws;
  float*    x     = (float*)w;    w += (size_t)BN * DD  * 4;   // fp32 residual
  _Float16* hbuf  = (_Float16*)w; w += (size_t)BN * KP1 * 2;   // LN out (f16)
  float*    qkv   = (float*)w;    w += (size_t)BN * 3 * DD * 4;
  _Float16* attnb = (_Float16*)w; w += (size_t)BN * KP1 * 2;   // attn out (f16)
  _Float16* ffb   = (_Float16*)w; w += (size_t)BN * FFF * 2;   // fc1 out (f16)
  _Float16* Qp    = (_Float16*)w; w += (size_t)BB * HH * NN * HDP * 2;
  _Float16* Kp    = (_Float16*)w; w += (size_t)BB * HH * NN * HDP * 2;
  _Float16* Vt    = (_Float16*)w; w += (size_t)BB * HH * NN * HDP * 2;
  _Float16* bpk   = (_Float16*)w;                              // packed B (f16)

  k_embed<<<4096, 256, 0, stream>>>(idx, tok, pos, x);

  dim3 gQKV(1216 / BNT, BN / BMT);    // N=1200 -> Npad 1216
  dim3 gPRJ(448 / BNT,  BN / BMT);    // N=400  -> Npad 448
  dim3 gFC1(1600 / BNT, BN / BMT);    // N=1600 exact
  dim3 gHEAD(VV / BNT,  BN / BMT);    // N=32000 exact
  int attnBlocks = (BB * HH * (NN / 16)) / 4;   // 512

  for (int l = 0; l < LL; ++l) {
    k_ln<<<BN, 256, 0, stream>>>(x, ln1w + l * DD, ln1b + l * DD, hbuf);
    k_pack_b<<<2048, 256, 0, stream>>>(qkvw + (size_t)l * DD * 3 * DD, bpk,
                                       DD, 3 * DD, KP1, 1216);
    k_gemm<0><<<gQKV, 256, 0, stream>>>(hbuf, KP1, bpk, KP1, qkvb + l * 3 * DD,
                                        nullptr, qkv, nullptr, 3 * DD, 13);
    k_qkv_split<<<4096, 256, 0, stream>>>(qkv, Qp, Kp, Vt);
    k_attn<<<attnBlocks, 128, 0, stream>>>(Qp, Kp, Vt, attnb);
    k_pack_b<<<2048, 256, 0, stream>>>(aow + (size_t)l * DD * DD, bpk,
                                       DD, DD, KP1, 448);
    k_gemm<1><<<gPRJ, 256, 0, stream>>>(attnb, KP1, bpk, KP1, aob + l * DD,
                                        x, x, nullptr, DD, 13);
    k_ln<<<BN, 256, 0, stream>>>(x, ln2w + l * DD, ln2b + l * DD, hbuf);
    k_pack_b<<<2048, 256, 0, stream>>>(fc1w + (size_t)l * DD * FFF, bpk,
                                       DD, FFF, KP1, FFF);
    k_gemm<2><<<gFC1, 256, 0, stream>>>(hbuf, KP1, bpk, KP1, fc1b + l * FFF,
                                        nullptr, nullptr, ffb, FFF, 13);
    k_pack_b<<<2048, 256, 0, stream>>>(fc2w + (size_t)l * FFF * DD, bpk,
                                       FFF, DD, FFF, 448);
    k_gemm<1><<<gPRJ, 256, 0, stream>>>(ffb, FFF, bpk, FFF, fc2b + l * DD,
                                        x, x, nullptr, DD, 50);
  }
  k_ln<<<BN, 256, 0, stream>>>(x, lnfw, lnfb, hbuf);
  k_pack_b<<<4096, 256, 0, stream>>>(headw, bpk, DD, VV, KP1, VV);
  k_gemm<0><<<gHEAD, 256, 0, stream>>>(hbuf, KP1, bpk, KP1, nullptr, nullptr,
                                       (float*)d_out, nullptr, VV, 13);
}
